// AttentionMechanism_68324339745373
// MI455X (gfx1250) — matrix operand order repaired and checked by reading, not compiled, not run on hardware
//
#include <hip/hip_runtime.h>

typedef __bf16 bf16_t;
typedef __attribute__((ext_vector_type(16))) __bf16 v16bf;
typedef __attribute__((ext_vector_type(8)))  __bf16 v8bf;
typedef __attribute__((ext_vector_type(8)))  float  v8f;

static __device__ __forceinline__ v8f wmma_bf16(v16bf a, v16bf b, v8f c) {
  // 8 args: (neg_a, A, neg_b, B, c_mod, C, reuse_a, reuse_b)
  return __builtin_amdgcn_wmma_f32_16x16x32_bf16(false, a, false, b, (short)0, c, false, false);
}

static __device__ __forceinline__ v16bf load32B(const bf16_t* p) {
  return *(const v16bf*)p;                     // 32B contiguous -> 2x b128
}
static __device__ __forceinline__ v16bf load16Bx2(const bf16_t* lo, const bf16_t* hi) {
  v8bf a = *(const v8bf*)lo;                   // 16B
  v8bf b = *(const v8bf*)hi;                   // 16B
  return __builtin_shufflevector(a, b, 0,1,2,3,4,5,6,7,8,9,10,11,12,13,14,15);
}

static __device__ __forceinline__ float fast_exp2(float x) {
#if __has_builtin(__builtin_amdgcn_exp2f)
  return __builtin_amdgcn_exp2f(x);            // v_exp_f32
#else
  return exp2f(x);
#endif
}

#define B_SZ   4
#define N_SZ   4096
#define C_SZ   256
#define D_SZ   32
#define ROWS   (B_SZ * N_SZ)              /* 16384 */
#define WPB    8                          /* waves per block */
#define TILES  (ROWS / 16)                /* 1024 16-row tiles */
#define BLOCKS (TILES / WPB)              /* 128 */
#define LOG2E  1.4426950408889634f
#define ESHIFT 16.0f                      /* global exp2 shift; cancels in softmax */

// ---------------------------------------------------------------------------
// Kernel 1: q/k = x @ W + b (row-major bf16 [ROWS,32]); v stored TRANSPOSED
// as vt[b][d][n]. Weights swizzled into LDS in B-fragment order.
// Wq/bq are pre-scaled by log2(e) so attention can use raw exp2.
// ---------------------------------------------------------------------------
__global__ __launch_bounds__(256) void sa_qkv(
    const float* __restrict__ x,
    const float* __restrict__ Wk, const float* __restrict__ bk,
    const float* __restrict__ Wq, const float* __restrict__ bq,
    const float* __restrict__ Wv, const float* __restrict__ bv,
    bf16_t* __restrict__ qb, bf16_t* __restrict__ kb, bf16_t* __restrict__ vt)
{
  // fragment-order weights: frag f = (m*8 + s)*2 + nh, 512 bf16 per frag
  __shared__ bf16_t ldsW[3 * 16 * 512];      // 48 KB
  const int tid = threadIdx.x;
  #pragma unroll
  for (int m = 0; m < 3; ++m) {
    const float* w = (m == 0) ? Wq : ((m == 1) ? Wk : Wv);
    const float scl = (m == 0) ? LOG2E : 1.0f;
    for (int i = tid; i < C_SZ * D_SZ; i += 256) {
      const int k = i >> 5, d = i & 31;                 // W[k][d], row-major
      const int s = k >> 5, kr = k & 31;
      const int nh = d >> 4, n = d & 15;
      const int ln = ((kr >> 4) << 4) | n;
      const int e  = kr & 15;
      const int f  = (m * 8 + s) * 2 + nh;
      ldsW[f * 512 + ln * 16 + e] = (bf16_t)(w[i] * scl);
    }
  }
  __syncthreads();

  const int lane  = tid & 31;
  const int wave  = tid >> 5;
  const int tile  = blockIdx.x * WPB + wave;
  const int g0    = tile * 16;

  const int mrow  = lane & 15;        // A row / C col
  const int khalf = lane >> 4;        // 0 or 1
  const int abase = khalf * 16;       // A-frag K base
  const int col   = lane & 15;
  const int rbase = khalf * 8;        // C/D row base

  v8f acc[3][2];
  #pragma unroll
  for (int m = 0; m < 3; ++m) { v8f z = {}; acc[m][0] = z; acc[m][1] = z; }

  const float* xrow = x + (size_t)(g0 + mrow) * C_SZ;
  #pragma unroll
  for (int s = 0; s < 8; ++s) {               // K loop over C=256 in steps of 32
    const int kc = s * 32;
    v8f xa = *(const v8f*)(xrow + kc + abase);        // 32B
    v8f xb = *(const v8f*)(xrow + kc + abase + 8);    // 32B
    v16bf a;
    #pragma unroll
    for (int e = 0; e < 8; ++e) { a[e] = (bf16_t)xa[e]; a[e + 8] = (bf16_t)xb[e]; }
    #pragma unroll
    for (int m = 0; m < 3; ++m) {
      #pragma unroll
      for (int nh = 0; nh < 2; ++nh) {
        const int f = (m * 8 + s) * 2 + nh;
        v16bf bf = load32B(&ldsW[f * 512 + lane * 16]);
        acc[m][nh] = wmma_bf16(a, bf, acc[m][nh]);
      }
    }
  }

  // epilogue: q,k row-major; v transposed (packed 16B store per lane)
  const int batch = g0 / N_SZ;
  const int n0    = g0 - batch * N_SZ;
  #pragma unroll
  for (int nh = 0; nh < 2; ++nh) {
    const int d = nh * 16 + col;
    const float bq_ = bq[d] * LOG2E, bk_ = bk[d], bv_ = bv[d];
    #pragma unroll
    for (int r = 0; r < 8; ++r) {
      qb[(size_t)(g0 + rbase + r) * D_SZ + d] = (bf16_t)(acc[0][nh][r] + bq_);
      kb[(size_t)(g0 + rbase + r) * D_SZ + d] = (bf16_t)(acc[1][nh][r] + bk_);
    }
    v8bf pk;
    #pragma unroll
    for (int r = 0; r < 8; ++r) pk[r] = (bf16_t)(acc[2][nh][r] + bv_);
    *(v8bf*)(vt + ((size_t)(batch * D_SZ + d)) * N_SZ + n0 + rbase) = pk;
  }
}

// ---------------------------------------------------------------------------
// Kernel 2: fused flash attention, 64-key blocks, max-free exact softmax:
//   softmax(s) = exp2(s*log2e - ESHIFT) / rowsum   (global shift cancels)
// Per-lane partial row sums; single cross-lane reduction at the end.
// ---------------------------------------------------------------------------
__global__ __launch_bounds__(256) void sa_attn(
    const bf16_t* __restrict__ qb, const bf16_t* __restrict__ kb,
    const bf16_t* __restrict__ vt, bf16_t* __restrict__ fb)
{
  __shared__ bf16_t ldsP[WPB][16 * 64];   // per-wave P tile (C->A transpose), 16 KB
  const int tid  = threadIdx.x;
  const int lane = tid & 31;
  const int wave = tid >> 5;
  bf16_t* Pw = ldsP[wave];

  const int tile  = blockIdx.x * WPB + wave;
  const int batch = tile / (N_SZ / 16);
  const int q0    = (tile % (N_SZ / 16)) * 16;
  const size_t base = (size_t)batch * N_SZ;

  const int mrow  = lane & 15;
  const int khalf = lane >> 4;
  const int abase = khalf * 16;
  const int bb    = khalf * 16;
  const int col   = lane & 15;
  const int rbase = khalf * 8;

  // Q A-fragment (held for the whole key loop; already scaled by log2e)
  const bf16_t* qrow = qb + (base + q0 + mrow) * D_SZ;
  v16bf qa = load16Bx2(qrow + abase, qrow + abase + 8);

  // per-lane fragment base pointers (fixed within the loop)
  const bf16_t* kbp  = kb + (base + col) * D_SZ + bb;                       // K^T frags
  const bf16_t* vtp0 = vt + ((size_t)(batch * D_SZ + col)) * N_SZ + bb;     // V^T, d half 0
  const bf16_t* vtp1 = vt + ((size_t)(batch * D_SZ + 16 + col)) * N_SZ + bb;// V^T, d half 1

  float lsum[8];
  v8f acc0 = {}, acc1 = {};
  #pragma unroll
  for (int r = 0; r < 8; ++r) lsum[r] = 0.0f;

  for (int kk0 = 0; kk0 < N_SZ; kk0 += 64) {
    // K^T B-fragments for 4 key tiles (contiguous 32B per lane)
    v16bf k0 = load32B(kbp + (size_t)(kk0 +  0) * D_SZ);
    v16bf k1 = load32B(kbp + (size_t)(kk0 + 16) * D_SZ);
    v16bf k2 = load32B(kbp + (size_t)(kk0 + 32) * D_SZ);
    v16bf k3 = load32B(kbp + (size_t)(kk0 + 48) * D_SZ);

    v8f cz = {};
    v8f s0 = wmma_bf16(qa, k0, cz);
    v8f s1 = wmma_bf16(qa, k1, cz);
    v8f s2 = wmma_bf16(qa, k2, cz);
    v8f s3 = wmma_bf16(qa, k3, cz);

    // p = exp2(s - shift); accumulate per-lane row sums; stage P (C->A layout)
    #pragma unroll
    for (int r = 0; r < 8; ++r) {
      const float p0 = fast_exp2(s0[r] - ESHIFT);
      const float p1 = fast_exp2(s1[r] - ESHIFT);
      const float p2 = fast_exp2(s2[r] - ESHIFT);
      const float p3 = fast_exp2(s3[r] - ESHIFT);
      lsum[r] += (p0 + p1) + (p2 + p3);
      bf16_t* prow = Pw + (rbase + r) * 64 + col;
      prow[0]  = (bf16_t)p0;
      prow[16] = (bf16_t)p1;
      prow[32] = (bf16_t)p2;
      prow[48] = (bf16_t)p3;
    }

    // P back as two A-fragments (contiguous 16B ds loads)
    const bf16_t* pr = Pw + mrow * 64;
    v16bf pa0 = load16Bx2(pr + abase,      pr + abase + 8);
    v16bf pa1 = load16Bx2(pr + 32 + abase, pr + 32 + abase + 8);

    // V B-fragments straight from transposed global V (contiguous 32B)
    v16bf v00 = load32B(vtp0 + kk0);
    v16bf v01 = load32B(vtp0 + kk0 + 32);
    v16bf v10 = load32B(vtp1 + kk0);
    v16bf v11 = load32B(vtp1 + kk0 + 32);

    acc0 = wmma_bf16(pa0, v00, acc0);
    acc0 = wmma_bf16(pa1, v01, acc0);
    acc1 = wmma_bf16(pa0, v10, acc1);
    acc1 = wmma_bf16(pa1, v11, acc1);
  }

  // one cross-lane reduction of the row sums, then normalize and store
  #pragma unroll
  for (int r = 0; r < 8; ++r) {
    float t = lsum[r];
    t += __shfl_xor(t, 1, 32);
    t += __shfl_xor(t, 2, 32);
    t += __shfl_xor(t, 4, 32);
    t += __shfl_xor(t, 8, 32);
    const float inv = 1.0f / t;
    const size_t row = base + q0 + rbase + r;
    fb[row * D_SZ + col]      = (bf16_t)(acc0[r] * inv);
    fb[row * D_SZ + 16 + col] = (bf16_t)(acc1[r] * inv);
  }
}

// ---------------------------------------------------------------------------
// Kernel 3: out = gamma * (feat @ Wo + bo) + x   -> f32 [ROWS, 256]
// Wo swizzled into LDS in B-fragment order.
// ---------------------------------------------------------------------------
__global__ __launch_bounds__(256) void sa_out(
    const bf16_t* __restrict__ fb, const float* __restrict__ Wo,
    const float* __restrict__ bo, const float* __restrict__ gamma,
    const float* __restrict__ x, float* __restrict__ out)
{
  __shared__ bf16_t ldsWo[16 * 512];   // 16 KB, fragment order (frag = col tile)
  const int tid = threadIdx.x;
  for (int i = tid; i < D_SZ * C_SZ; i += 256) {
    const int k = i >> 8, c = i & 255;                 // Wo[k][c], row-major
    const int ct = c >> 4, n = c & 15;
    const int ln = ((k >> 4) << 4) | n;
    const int e  = k & 15;
    ldsWo[ct * 512 + ln * 16 + e] = (bf16_t)Wo[i];
  }
  __syncthreads();

  const int lane  = tid & 31;
  const int wave  = tid >> 5;
  const int tile  = blockIdx.x * WPB + wave;
  const int g0    = tile * 16;

  const int mrow  = lane & 15;
  const int khalf = lane >> 4;
  const int abase = khalf * 16;
  const int col   = lane & 15;
  const int rbase = khalf * 8;

  const bf16_t* frow = fb + (size_t)(g0 + mrow) * D_SZ;
  v16bf a = load16Bx2(frow + abase, frow + abase + 8);

  const float g = gamma[0];

  #pragma unroll 4
  for (int ct = 0; ct < 16; ++ct) {          // 256 output cols in 16-wide tiles
    const int c = ct * 16 + col;
    v16bf b = load32B(&ldsWo[ct * 512 + lane * 16]);
    v8f cz = {};
    v8f o = wmma_bf16(a, b, cz);
    const float bias = bo[c];
    #pragma unroll
    for (int r = 0; r < 8; ++r) {
      const size_t row = (size_t)(g0 + rbase + r);
      out[row * C_SZ + c] = g * (o[r] + bias) + x[row * C_SZ + c];
    }
  }
}

// ---------------------------------------------------------------------------
extern "C" void kernel_launch(void* const* d_in, const int* in_sizes, int n_in,
                              void* d_out, int out_size, void* d_ws, size_t ws_size,
                              hipStream_t stream) {
  const float* x     = (const float*)d_in[0];
  const float* Wk    = (const float*)d_in[1];
  const float* bk    = (const float*)d_in[2];
  const float* Wq    = (const float*)d_in[3];
  const float* bq    = (const float*)d_in[4];
  const float* Wv    = (const float*)d_in[5];
  const float* bv    = (const float*)d_in[6];
  const float* Wo    = (const float*)d_in[7];
  const float* bo    = (const float*)d_in[8];
  const float* gamma = (const float*)d_in[9];
  float* out = (float*)d_out;

  // workspace: q, k row-major; v transposed [B][D][N]; feat row-major (1 MB each)
  bf16_t* qb = (bf16_t*)d_ws;
  bf16_t* kb = qb + (size_t)ROWS * D_SZ;
  bf16_t* vt = kb + (size_t)ROWS * D_SZ;
  bf16_t* fb = vt + (size_t)B_SZ * D_SZ * N_SZ;

  dim3 grid(BLOCKS), block(256);
  sa_qkv <<<grid, block, 0, stream>>>(x, Wk, bk, Wq, bq, Wv, bv, qb, kb, vt);
  sa_attn<<<grid, block, 0, stream>>>(qb, kb, vt, fb);
  sa_out <<<grid, block, 0, stream>>>(fb, Wo, bo, gamma, x, out);
}
